// FlowGuidedAlign_26723286516110
// MI455X (gfx1250) — matrix operand order repaired and checked by reading, not compiled, not run on hardware
//
#include <hip/hip_runtime.h>
#include <hip/hip_bf16.h>

typedef __attribute__((ext_vector_type(16))) __bf16 v16bf;
typedef __attribute__((ext_vector_type(8)))  __bf16 v8bf;
typedef __attribute__((ext_vector_type(4)))  __bf16 v4bf;
typedef __attribute__((ext_vector_type(8)))  float  v8f;

#define NFEAT 64
#define GRP   8
#define KK9   9
#define BSZ   2
#define HH    128
#define WW    128
#define HWPIX (HH*WW)
#define NPIX  (BSZ*HWPIX)   // 32768 GEMM columns

__device__ __forceinline__ float lrelu_f(float x){ return x >= 0.f ? x : 0.1f*x; }

// ---------------- packing / conversion kernels ----------------

// Pack an f32 weight matrix [M, Kreal] into bf16 [M, Kpad] (zero pad K tail).
__global__ void pack_weight_bf16(const float* __restrict__ src, __bf16* __restrict__ dst,
                                 int M, int Kreal, int Kpad){
  int i = blockIdx.x*blockDim.x + threadIdx.x;
  int tot = M*Kpad;
  if(i >= tot) return;
  int m = i / Kpad, k = i - m*Kpad;
  dst[i] = (k < Kreal) ? (__bf16)src[(size_t)m*Kreal + k] : (__bf16)0.0f;
}

// Build concatenated conv1 input [B, 130, H, W] in bf16.
__global__ void build_x0(const float* __restrict__ fflow, const float* __restrict__ shake,
                         const float* __restrict__ flows, __bf16* __restrict__ x0){
  int i = blockIdx.x*blockDim.x + threadIdx.x;
  const int C = 2*NFEAT + 2;
  int tot = BSZ * C * HWPIX;
  if(i >= tot) return;
  int p = i % HWPIX; int c = (i / HWPIX) % C; int b = i / (HWPIX*C);
  float v;
  if(c < NFEAT)        v = fflow[((size_t)b*NFEAT + c)*HWPIX + p];
  else if(c < 2*NFEAT) v = shake[((size_t)b*NFEAT + (c-NFEAT))*HWPIX + p];
  else                 v = flows[((size_t)b*2 + (c-2*NFEAT))*HWPIX + p];
  x0[i] = (__bf16)v;
}

// 3x3 pad-1 im2col: src [B,Cin,H,W] bf16 -> col [NPIX, Kpad] bf16, K-major rows.
__global__ void im2col3x3(const __bf16* __restrict__ src, __bf16* __restrict__ col,
                          int Cin, int Kpad){
  int i = blockIdx.x*blockDim.x + threadIdx.x;
  int tot = NPIX * Cin;
  if(i >= tot) return;
  int c = i % Cin;
  int n = i / Cin;
  int b = n / HWPIX, p = n - b*HWPIX;
  int h = p / WW,    w = p - h*WW;
  const __bf16* s = src + (size_t)(b*Cin + c)*HWPIX;
  __bf16* d = col + (size_t)n*Kpad + c*KK9;
  #pragma unroll
  for(int kh=0;kh<3;++kh){
    int y = h + kh - 1;
    #pragma unroll
    for(int kw=0;kw<3;++kw){
      int x = w + kw - 1;
      __bf16 v = (__bf16)0.0f;
      if(y >= 0 && y < HH && x >= 0 && x < WW) v = s[y*WW + x];
      d[kh*3+kw] = v;
    }
  }
  if(c == Cin-1){
    for(int k = Cin*KK9; k < Kpad; ++k) col[(size_t)n*Kpad + k] = (__bf16)0.0f;
  }
}

// ---------------- WMMA GEMM over K-major columns ----------------
// out[b,oc,p] = act( bias[oc] + sum_k A[oc,k] * col[n][k] ),  n = b*HW+p
// Block = 128 threads = 4 waves, sharing one 16-row A tile staged whole in LDS
// (one barrier total). Each wave owns FOUR N-tiles: one A fragment per K-step
// feeds 4 WMMAs. Grid divides exactly -> no early exits, EXEC all ones.
template<int ACT, int KPAD>
__global__ void gemm_wmma_col(const __bf16* __restrict__ Ap, const __bf16* __restrict__ Bcol,
                              const float* __restrict__ bias,
                              float* __restrict__ outF, __bf16* __restrict__ outH,
                              int Cout, int mTiles){
  __shared__ __align__(32) __bf16 As[16*KPAD];
  const int tid  = threadIdx.x;
  const int lane = tid & 31;
  const int wv   = tid >> 5;                   // 0..3
  const int mT   = blockIdx.x % mTiles;
  const int nTB  = blockIdx.x / mTiles;        // covers 16 N-tiles
  const int oc0  = mT * 16;
  const int half = lane >> 4;
  const int mrow = lane & 15;
  const int colb = (nTB*16 + 4*wv)*16 + (lane & 15);
  const __bf16* brow0 = Bcol + (size_t)(colb     ) * KPAD;
  const __bf16* brow1 = Bcol + (size_t)(colb + 16) * KPAD;
  const __bf16* brow2 = Bcol + (size_t)(colb + 32) * KPAD;
  const __bf16* brow3 = Bcol + (size_t)(colb + 48) * KPAD;

  // ---- stage the whole 16 x KPAD A tile once ----
  {
    const int sr  = tid >> 3;                  // 0..15 row
    const int skb = (tid & 7) * 4;             // 0..28
    const bool svalid = (oc0 + sr) < Cout;
    const __bf16* g = Ap + (size_t)(oc0 + sr) * KPAD + skb;
    __bf16* l = As + sr*KPAD + skb;
    #pragma unroll 4
    for(int k0 = 0; k0 < KPAD; k0 += 32){
      v4bf av = {};
      if(svalid) av = *(const v4bf*)(g + k0);
      *(v4bf*)(l + k0) = av;
    }
  }
  __syncthreads();

  v8f acc0 = {}, acc1 = {}, acc2 = {}, acc3 = {};
  const __bf16* arowl = As + mrow*KPAD;
  for(int k0 = 0; k0 < KPAD; k0 += 32){
    // A fragment (contiguous K, matching B):
    // half=0: K = {0..15}; half=1: K = {16..31}
    v16bf a;
    {
      v8bf lo = *(const v8bf*)(arowl + k0 + half*16);
      v8bf hi = *(const v8bf*)(arowl + k0 + half*16 + 8);
      #pragma unroll
      for(int e=0;e<8;++e){ a[e] = lo[e]; a[e+8] = hi[e]; }
    }
    __builtin_prefetch(brow0 + k0 + 256, 0, 0);
    // B fragments (32x16): lane column fixed, K = k0 + half*16 + e (contiguous)
    v16bf b0 = *(const v16bf*)(brow0 + k0 + half*16);
    v16bf b1 = *(const v16bf*)(brow1 + k0 + half*16);
    v16bf b2 = *(const v16bf*)(brow2 + k0 + half*16);
    v16bf b3 = *(const v16bf*)(brow3 + k0 + half*16);
    acc0 = __builtin_amdgcn_wmma_f32_16x16x32_bf16(false,a,false,b0,(short)0,acc0,false,false);
    acc1 = __builtin_amdgcn_wmma_f32_16x16x32_bf16(false,a,false,b1,(short)0,acc1,false,false);
    acc2 = __builtin_amdgcn_wmma_f32_16x16x32_bf16(false,a,false,b2,(short)0,acc2,false,false);
    acc3 = __builtin_amdgcn_wmma_f32_16x16x32_bf16(false,a,false,b3,(short)0,acc3,false,false);
  }

  // ---- C store: VGPR r, half -> M = r + 8*half; column = colb + 16*j ----
  #pragma unroll
  for(int r=0;r<8;++r){
    int oc = oc0 + r + 8*half;
    if(oc >= Cout) continue;
    float bz = bias[oc];
    float vv[4] = { acc0[r]+bz, acc1[r]+bz, acc2[r]+bz, acc3[r]+bz };
    #pragma unroll
    for(int j=0;j<4;++j){
      float v = vv[j];
      if(ACT == 1) v = lrelu_f(v);
      int cn = colb + 16*j;
      int bi = cn / HWPIX, pi = cn - bi*HWPIX;
      size_t o = ((size_t)bi*Cout + oc)*HWPIX + pi;
      if(outF) outF[o] = v;
      if(outH) outH[o] = (__bf16)v;
    }
  }
}

// ---------------- modulated deformable sampling ----------------
// V[n][(g*8+c)*9+kk] = mask * bilinear(feat[b, g*8+c], h-1+kh+dy, w-1+kw+dx)
__global__ void dcn_sample(const float* __restrict__ feat,  // [B,64,H,W]
                           const float* __restrict__ om,    // [B,216,HW] f32
                           const float* __restrict__ flows, // [B,2,HW]
                           __bf16* __restrict__ Vt){        // [NPIX, 576]
  int i = blockIdx.x*blockDim.x + threadIdx.x;
  const int tot = BSZ * GRP * KK9 * HWPIX;
  if(i >= tot) return;
  int p  = i % HWPIX;
  int kk = (i / HWPIX) % KK9;
  int g  = (i / (HWPIX*KK9)) % GRP;
  int b  =  i / (HWPIX*KK9*GRP);
  int h = p / WW, w = p - (p / WW)*WW;

  const float* omb = om + (size_t)b*216*HWPIX;
  int oc = (g*KK9 + kk)*2;                      // offset chan maps to om chan directly
  float dy = omb[(size_t)oc*HWPIX + p]     + flows[((size_t)b*2 + 1)*HWPIX + p];
  float dx = omb[(size_t)(oc+1)*HWPIX + p] + flows[((size_t)b*2 + 0)*HWPIX + p];
  float mk = omb[(size_t)(144 + g*KK9 + kk)*HWPIX + p];
  mk = 1.0f / (1.0f + expf(-mk));

  int kh = kk / 3, kw = kk - kh*3;
  float y = (float)h - 1.0f + (float)kh + dy;
  float x = (float)w - 1.0f + (float)kw + dx;
  float y0f = floorf(y), x0f = floorf(x);
  float fy = y - y0f, fx = x - x0f;
  int y0 = (int)y0f, x0 = (int)x0f;
  int y1 = y0 + 1,   x1 = x0 + 1;
  float w00 = (1.f-fy)*(1.f-fx), w01 = (1.f-fy)*fx;
  float w10 = fy*(1.f-fx),       w11 = fy*fx;
  bool vy0 = (y0>=0)&&(y0<HH), vy1 = (y1>=0)&&(y1<HH);
  bool vx0 = (x0>=0)&&(x0<WW), vx1 = (x1>=0)&&(x1<WW);
  int cy0 = y0<0?0:(y0>HH-1?HH-1:y0), cy1 = y1<0?0:(y1>HH-1?HH-1:y1);
  int cx0 = x0<0?0:(x0>WW-1?WW-1:x0), cx1 = x1<0?0:(x1>WW-1?WW-1:x1);
  w00 = (vy0&&vx0)? w00 : 0.f;  w01 = (vy0&&vx1)? w01 : 0.f;
  w10 = (vy1&&vx0)? w10 : 0.f;  w11 = (vy1&&vx1)? w11 : 0.f;
  int i00 = cy0*WW+cx0, i01 = cy0*WW+cx1, i10 = cy1*WW+cx0, i11 = cy1*WW+cx1;

  __bf16* vrow = Vt + (size_t)(b*HWPIX + p)*(NFEAT*KK9);
  const float* fb = feat + ((size_t)b*NFEAT + g*8)*HWPIX;
  #pragma unroll
  for(int c=0;c<8;++c){
    const float* fc = fb + (size_t)c*HWPIX;
    float sv = fc[i00]*w00 + fc[i01]*w01 + fc[i10]*w10 + fc[i11]*w11;
    vrow[(g*8 + c)*KK9 + kk] = (__bf16)(sv * mk);
  }
}

// ---------------- host side ----------------

static inline int cdiv_i(int a, int b){ return (a + b - 1) / b; }

extern "C" void kernel_launch(void* const* d_in, const int* in_sizes, int n_in,
                              void* d_out, int out_size, void* d_ws, size_t ws_size,
                              hipStream_t stream) {
  (void)in_sizes; (void)n_in; (void)out_size; (void)ws_size;
  const float* ref_fea  = (const float*)d_in[0];   // [B,64,H,W]
  const float* ref_flow = (const float*)d_in[1];   // [B,64,H,W]
  const float* shake    = (const float*)d_in[2];   // [B,64,H,W]
  const float* flows    = (const float*)d_in[3];   // [B,2,H,W]
  const float* w1  = (const float*)d_in[4];  const float* b1  = (const float*)d_in[5];
  const float* w2  = (const float*)d_in[6];  const float* b2  = (const float*)d_in[7];
  const float* wom = (const float*)d_in[8];  const float* bom = (const float*)d_in[9];
  const float* wd  = (const float*)d_in[10]; const float* bd  = (const float*)d_in[11];
  float* out = (float*)d_out;

  const int C0  = 2*NFEAT + 2;     // 130
  const int K1  = C0*KK9;          // 1170
  const int K1p = 1184;            // pad to multiple of 32
  const int K2  = NFEAT*KK9;       // 576 (already multiple of 32)
  const int COM = 3*GRP*KK9;       // 216

  // carve workspace (256B aligned slabs)
  size_t off = 0;
  char* base = (char*)d_ws;
  auto carve = [&](size_t bytes)->void*{
    void* pp = base + off;
    off += (bytes + 255) & ~(size_t)255;
    return pp;
  };
  __bf16* X0   = (__bf16*)carve((size_t)BSZ*C0*HWPIX*sizeof(__bf16));
  __bf16* W1h  = (__bf16*)carve((size_t)NFEAT*K1p*sizeof(__bf16));
  __bf16* F1   = (__bf16*)carve((size_t)BSZ*NFEAT*HWPIX*sizeof(__bf16));
  __bf16* W2h  = (__bf16*)carve((size_t)NFEAT*K2*sizeof(__bf16));
  __bf16* F2   = (__bf16*)carve((size_t)BSZ*NFEAT*HWPIX*sizeof(__bf16));
  __bf16* WOMh = (__bf16*)carve((size_t)COM*K2*sizeof(__bf16));
  float*  OM   = (float*) carve((size_t)BSZ*COM*HWPIX*sizeof(float));
  __bf16* WDh  = (__bf16*)carve((size_t)NFEAT*K2*sizeof(__bf16));
  // one big K-major scratch, reused sequentially: col1 -> col2 -> col_om -> V
  __bf16* COL  = (__bf16*)carve((size_t)NPIX*K1p*sizeof(__bf16));

  // pack weights / inputs to bf16
  pack_weight_bf16<<<cdiv_i(NFEAT*K1p,256),256,0,stream>>>(w1,  W1h, NFEAT, K1, K1p);
  pack_weight_bf16<<<cdiv_i(NFEAT*K2 ,256),256,0,stream>>>(w2,  W2h, NFEAT, K2, K2);
  pack_weight_bf16<<<cdiv_i(COM*K2   ,256),256,0,stream>>>(wom, WOMh, COM,  K2, K2);
  pack_weight_bf16<<<cdiv_i(NFEAT*K2 ,256),256,0,stream>>>(wd,  WDh, NFEAT, K2, K2);
  build_x0<<<cdiv_i(BSZ*C0*HWPIX,256),256,0,stream>>>(ref_flow, shake, flows, X0);

  const int nTiles = NPIX/16;          // 2048, divisible by 16
  const int nBlkN  = nTiles/16;        // N-tile blocks per M-tile (128)

  // conv1: 130 -> 64, lrelu, bf16 out
  im2col3x3<<<cdiv_i(NPIX*C0,256),256,0,stream>>>(X0, COL, C0, K1p);
  {
    int mT = cdiv_i(NFEAT,16);
    gemm_wmma_col<1,1184><<<mT*nBlkN,128,0,stream>>>(W1h, COL, b1, nullptr, F1,
                                                     NFEAT, mT);
  }
  // conv2: 64 -> 64, lrelu, bf16 out
  im2col3x3<<<cdiv_i(NPIX*NFEAT,256),256,0,stream>>>(F1, COL, NFEAT, K2);
  {
    int mT = cdiv_i(NFEAT,16);
    gemm_wmma_col<1,576><<<mT*nBlkN,128,0,stream>>>(W2h, COL, b2, nullptr, F2,
                                                    NFEAT, mT);
  }
  // conv_om: 64 -> 216, no act, f32 out
  im2col3x3<<<cdiv_i(NPIX*NFEAT,256),256,0,stream>>>(F2, COL, NFEAT, K2);
  {
    int mT = cdiv_i(COM,16);
    gemm_wmma_col<0,576><<<mT*nBlkN,128,0,stream>>>(WOMh, COL, bom, OM, nullptr,
                                                    COM, mT);
  }
  // deformable sampling -> V [NPIX, 576] bf16 (reuses COL)
  dcn_sample<<<cdiv_i(BSZ*GRP*KK9*HWPIX,256),256,0,stream>>>(ref_fea, OM, flows, COL);

  // final einsum GEMM: 64 x 576 x NPIX, lrelu, f32 out to d_out
  {
    int mT = cdiv_i(NFEAT,16);
    gemm_wmma_col<1,576><<<mT*nBlkN,128,0,stream>>>(WDh, COL, bd, out, nullptr,
                                                    NFEAT, mT);
  }
}
